// enhance_net_nopool_54863912240030
// MI455X (gfx1250) — matrix operand order repaired and checked by reading, not compiled, not run on hardware
//
#include <hip/hip_runtime.h>
#include <math.h>

typedef __attribute__((ext_vector_type(16))) _Float16 v16h;
typedef __attribute__((ext_vector_type(8)))  float    v8f;

// ---------------- problem constants ----------------
static constexpr int BB = 16;
static constexpr int HH = 512;
static constexpr int WWID = 512;
static constexpr int HWP = HH * WWID;          // 262144 = 2^18
// fused-tile constants (patch: y-halo 7, x-halo 8 so rows are 16B aligned)
static constexpr int TILE = 32;
static constexpr int HALOY = 7;
static constexpr int HALOX = 8;
static constexpr int PATCH_Y = 46;             // TILE + 2*HALOY
static constexpr int PATCH_X = 48;             // TILE + 2*HALOX
static constexpr int PSTR = 48;                // row stride (halfs) == PATCH_X
static constexpr int PP = PSTR * PSTR;         // 2304 halfs per channel plane
static constexpr int NPLANES = 39;             // v + 9*4ch + zero plane + trash plane
static constexpr int ZPLANE = 37;              // all-zero plane (read-only pad)
static constexpr int TPLANE = 38;              // trash plane (write-only sink)
static constexpr int ACT_HALFS = PP * NPLANES; // 89856
static constexpr int WTOT2 = 14848;            // padded weights: sum 16*Kpad
static constexpr int BLDS_BYTE_OFF = 2 * (ACT_HALFS + WTOT2); // 209408
static constexpr int RLDS_BYTE_OFF = BLDS_BYTE_OFF + 40 * 4;  // 209568 (16B aligned)
static constexpr size_t SMEM_BYTES = (size_t)RLDS_BYTE_OFF + (TILE * TILE + 1) * 4; // 213668

// ------------- compile-time per-conv metadata (conv cv -> tensor id cv) ------
constexpr int K_CIN[10]   = {1, 4, 8, 4, 8, 12, 4, 8, 12, 16};
constexpr int K_COUT[10]  = {4, 4, 4, 4, 4, 4, 4, 4, 4, 1};
constexpr int K_DEP[10]   = {1, 2, 3, 3, 4, 5, 4, 5, 6, 7};
constexpr int K_WOFF2[10] = {0, 512, 1536, 3072, 4096, 5632, 7680, 8704, 10240, 12288};
// input-channel -> LDS plane (padded channels -> zero plane 37)
constexpr int K_CHB[10][20] = {
  {0,37,37,37,37,37,37,37,37,37,37,37,37,37,37,37,37,37,37,37},
  {1,2,3,4,37,37,37,37,37,37,37,37,37,37,37,37,37,37,37,37},
  {1,2,3,4,5,6,7,8,37,37,37,37,37,37,37,37,37,37,37,37},
  {5,6,7,8,37,37,37,37,37,37,37,37,37,37,37,37,37,37,37,37},
  {5,6,7,8,13,14,15,16,37,37,37,37,37,37,37,37,37,37,37,37},
  {1,2,3,4,9,10,11,12,17,18,19,20,37,37,37,37,37,37,37,37},
  {13,14,15,16,37,37,37,37,37,37,37,37,37,37,37,37,37,37,37,37},
  {13,14,15,16,25,26,27,28,37,37,37,37,37,37,37,37,37,37,37,37},
  {5,6,7,8,17,18,19,20,29,30,31,32,37,37,37,37,37,37,37,37},
  {1,2,3,4,9,10,11,12,21,22,23,24,33,34,35,36,37,37,37,37}
};
// LDS half-offset of A element for global K index k (plane + 3x3 tap), constexpr
constexpr int a_off(int cv, int k) {
  int c = k / 9;
  int t = k - c * 9;
  return K_CHB[cv][c] * PP + (t / 3) * PSTR + (t % 3);
}

// ---------------- kernel 1: v = (r+g+b)/3 ----------------
__global__ __launch_bounds__(256) void prep_v_kernel(
    const float* __restrict__ x, float* __restrict__ vf, _Float16* __restrict__ vh) {
  size_t p = (size_t)blockIdx.x * 256 + threadIdx.x;   // over B*HW
  int b = (int)(p >> 18);
  size_t rem = p & (size_t)(HWP - 1);
  size_t base = (size_t)b * 3 * HWP + rem;
  float v = (x[base] + x[base + HWP] + x[base + 2 * HWP]) * (1.0f / 3.0f);
  vf[p] = v;
  vh[p] = (_Float16)v;
}

// ---------- fully specialized conv level: all shape math constant-folds ------
template <int CV>
__device__ __forceinline__ void conv_level(
    _Float16* __restrict__ smh, const _Float16* __restrict__ wlds2,
    const float* __restrict__ blds, float* __restrict__ rlds,
    int wave, int n, int hgrp, int gx0, int gy0) {
  constexpr int d    = K_DEP[CV];
  constexpr int Sy   = PATCH_Y - 2 * d;
  constexpr int Sx   = PATCH_X - 2 * d;
  constexpr int mtot = Sy * Sx;
  constexpr int Cin  = K_CIN[CV];
  constexpr int K    = Cin * 9;
  constexpr int nK   = (K + 31) / 32;
  constexpr int Kpad = nK * 32;
  constexpr int nM   = (mtot + 15) / 16;
  constexpr int co   = K_COUT[CV];
  constexpr int outbase = (1 + 4 * CV) * PP;

  const _Float16* brow = wlds2 + K_WOFF2[CV] + n * Kpad + hgrp * 16;
  const float bias = (n < co) ? blds[CV * 4 + n] : 0.f;
  const bool ncol = (n < co);

  for (int mt = wave; mt < nM; mt += 8) {
    v8f acc = {};
    // A row for this lane (clamped; duplicate rows discarded in epilogue)
    int m = mt * 16 + n;
    if (m >= mtot) m = mtot - 1;
    int q = m / Sx;                      // constant divisor -> magic multiply
    int pixb = (d + q - 1) * PSTR + (d + (m - q * Sx) - 1);

#pragma unroll
    for (int kc = 0; kc < nK; ++kc) {
      v16h a;
#pragma unroll
      for (int i = 0; i < 16; ++i) {
        // k for hgrp=0 / hgrp=1 are both compile-time constants
        const int k0 = kc * 32 + i;
        const int o0 = a_off(CV, k0);
        const int o1 = a_off(CV, k0 + 16);
        a[i] = smh[pixb + o0 + hgrp * (o1 - o0)];
      }
      v16h bfr = *(const v16h*)(brow + kc * 32);   // 32B aligned, zero-padded
      acc = __builtin_amdgcn_wmma_f32_16x16x32_f16(
          false, a, false, bfr, (short)0, acc, false, false);
    }

    // branch-free epilogue: invalid lanes write to the trash plane/slot
    int mm0 = mt * 16 + hgrp * 8;
    int q0 = mm0 / Sx;
    int p0 = mm0 - q0 * Sx;
#pragma unroll
    for (int rr = 0; rr < 8; ++rr) {
      int mm = mm0 + rr;
      int wr = (p0 + rr >= Sx) ? 1 : 0;
      int py2 = d + q0 + wr;
      int px2 = d + p0 + rr - wr * Sx;
      float val = acc[rr] + bias;
      if constexpr (CV < 9) {
        bool valid = (mm < mtot) & ncol;
        int gy = gy0 + py2, gx = gx0 + px2;
        bool inimg = (gy >= 0 && gy < HH && gx >= 0 && gx < WWID);
        val = val > 0.f ? val : 0.f;                      // relu
        _Float16 hv = inimg ? (_Float16)val : (_Float16)0.f;  // zero outside image
        int addr = valid ? (outbase + n * PP + py2 * PSTR + px2) : (TPLANE * PP + n);
        smh[addr] = hv;
      } else {
        // r conv: sigmoid -> LDS tile (owned pixels only), async-stored later
        bool valid = (mm < mtot) & (n == 0) &
                     (px2 >= HALOX) & (px2 < HALOX + TILE);
        float sg = 1.f / (1.f + expf(-val));              // sigmoid
        int ridx = valid ? ((py2 - HALOY) * TILE + (px2 - HALOX)) : (TILE * TILE);
        rlds[ridx] = sg;
      }
    }
  }
}

// ---------------- kernel 2: fused UNet++ tile with WMMA ----------------
__global__ __launch_bounds__(256) void unet_tile_kernel(
    const _Float16* __restrict__ v16,
    const float* w00, const float* b00, const float* w10, const float* b10,
    const float* w01, const float* b01, const float* w20, const float* b20,
    const float* w11, const float* b11, const float* w02, const float* b02,
    const float* w30, const float* b30, const float* w21, const float* b21,
    const float* w12, const float* b12, const float* w03, const float* b03,
    float* __restrict__ r_out) {
  extern __shared__ char smem_raw[];
  _Float16* smh   = (_Float16*)smem_raw;                 // 39 planes of PP halfs
  _Float16* wlds2 = smh + ACT_HALFS;                     // padded f16 weights
  float*    blds  = (float*)(smem_raw + BLDS_BYTE_OFF);  // 40 biases
  float*    rlds  = (float*)(smem_raw + RLDS_BYTE_OFF);  // 32x32 r tile + trash

  const int tid  = threadIdx.x;
  const int wave = tid >> 5;
  const int lane = tid & 31;
  const int n    = lane & 15;     // WMMA column (output channel)
  const int hgrp = lane >> 4;     // lane half-group

  const int tx = blockIdx.x, ty = blockIdx.y, b = blockIdx.z;
  const int gx0 = tx * TILE - HALOX;
  const int gy0 = ty * TILE - HALOY;

  // ---- stage padded f16 weights (zero-filled) + biases into LDS ----
  {
    const float* wsrc[10] = {w00, w10, w01, w20, w11, w02, w30, w21, w12, w03};
    const float* bsrc[10] = {b00, b10, b01, b20, b11, b02, b30, b21, b12, b03};
#pragma unroll
    for (int cv = 0; cv < 10; ++cv) {
      const int K = K_CIN[cv] * 9;
      const int Kpad = ((K + 31) >> 5) << 5;
      const int co = K_COUT[cv], off = K_WOFF2[cv];
      const int tot = 16 * Kpad;
      const float* wp = wsrc[cv];
      for (int i = tid; i < tot; i += 256) {
        int nn = i / Kpad, k = i - nn * Kpad;
        _Float16 val = (_Float16)0.f;
        if (nn < co && k < K) val = (_Float16)wp[nn * K + k];
        wlds2[off + i] = val;
      }
      if (tid < co) blds[cv * 4 + tid] = bsrc[cv][tid];
    }
  }
  // ---- zero plane ----
  for (int i = tid; i < PP; i += 256) smh[ZPLANE * PP + i] = (_Float16)0.f;

  // ---- stage v patch (46 rows x 48 halfs); async DMA for interior tiles ----
  const bool interior = (tx >= 1 && tx <= 14 && ty >= 1 && ty <= 14);
  if (interior) {
    // 46 rows * 96B = 276 lane-transfers of 16B each, all 16B aligned
    unsigned gbase = (unsigned)(((size_t)b * HWP + (size_t)gy0 * WWID + gx0) * 2);
    unsigned lbase = (unsigned)(size_t)smh;   // low 32 bits = LDS byte offset
    for (int t = tid; t < 276; t += 256) {
      int row = t / 6, seg = t - row * 6;
      unsigned goff = gbase + (unsigned)(row * (WWID * 2) + seg * 16);
      unsigned loff = lbase + (unsigned)(row * (PSTR * 2) + seg * 16);
      asm volatile("global_load_async_to_lds_b128 %0, %1, %2"
                   :: "v"(loff), "v"(goff), "s"(v16) : "memory");
    }
    asm volatile("s_wait_asynccnt 0" ::: "memory");
  } else {
    for (int i = tid; i < PATCH_Y * PATCH_X; i += 256) {
      int py = i / PATCH_X, px = i - py * PATCH_X;
      int gy = gy0 + py, gx = gx0 + px;
      _Float16 val = (_Float16)0.f;
      if (gy >= 0 && gy < HH && gx >= 0 && gx < WWID)
        val = v16[(size_t)b * HWP + (size_t)gy * WWID + gx];
      smh[py * PSTR + px] = val;
    }
  }
  __syncthreads();

  // ---- 10 conv levels, fully specialized ----
  conv_level<0>(smh, wlds2, blds, rlds, wave, n, hgrp, gx0, gy0);
  __syncthreads();
  conv_level<1>(smh, wlds2, blds, rlds, wave, n, hgrp, gx0, gy0);
  __syncthreads();
  conv_level<2>(smh, wlds2, blds, rlds, wave, n, hgrp, gx0, gy0);
  __syncthreads();
  conv_level<3>(smh, wlds2, blds, rlds, wave, n, hgrp, gx0, gy0);
  __syncthreads();
  conv_level<4>(smh, wlds2, blds, rlds, wave, n, hgrp, gx0, gy0);
  __syncthreads();
  conv_level<5>(smh, wlds2, blds, rlds, wave, n, hgrp, gx0, gy0);
  __syncthreads();
  conv_level<6>(smh, wlds2, blds, rlds, wave, n, hgrp, gx0, gy0);
  __syncthreads();
  conv_level<7>(smh, wlds2, blds, rlds, wave, n, hgrp, gx0, gy0);
  __syncthreads();
  conv_level<8>(smh, wlds2, blds, rlds, wave, n, hgrp, gx0, gy0);
  __syncthreads();
  conv_level<9>(smh, wlds2, blds, rlds, wave, n, hgrp, gx0, gy0);
  __syncthreads();

  // ---- async store of the 32x32 f32 r tile (32 rows x 8 x 16B, all aligned) --
  {
    int row = tid >> 3, seg = tid & 7;
    unsigned gbase = (unsigned)(((size_t)b * HWP + (size_t)(ty * TILE) * WWID
                                 + tx * TILE) * 4);
    unsigned goff = gbase + (unsigned)(row * (WWID * 4) + seg * 16);
    unsigned loff = (unsigned)(size_t)rlds + (unsigned)(row * (TILE * 4) + seg * 16);
    asm volatile("global_store_async_from_lds_b128 %0, %1, %2"
                 :: "v"(goff), "v"(loff), "s"(r_out) : "memory");
    asm volatile("s_wait_asynccnt 0" ::: "memory");
  }
}

// ------- kernel 3: nearest32 -> stride-2 conv -> lrelu -> instance norm -------
__global__ __launch_bounds__(256) void smallmap_kernel(
    const float* __restrict__ src, const float* __restrict__ w,
    const float* __restrict__ bias, const float* __restrict__ g,
    const float* __restrict__ bt, float* __restrict__ out) {
  const int b = blockIdx.x;
  const int p = threadIdx.x;           // 256 = 16x16 output pixels
  __shared__ float in32[32][32];
  __shared__ float t[4][256];
  __shared__ float red[256];

  for (int i = p; i < 1024; i += 256) {
    int iy = i >> 5, ix = i & 31;
    in32[iy][ix] = src[(size_t)b * HWP + (size_t)(iy * 16) * WWID + ix * 16];
  }
  __syncthreads();

  const int oy = p >> 4, ox = p & 15;
  for (int cc = 0; cc < 4; ++cc) {
    float acc = bias[cc];
#pragma unroll
    for (int ky = 0; ky < 3; ++ky)
#pragma unroll
      for (int kx = 0; kx < 3; ++kx) {
        int y = 2 * oy + ky - 1, xx = 2 * ox + kx - 1;
        float vv = (y >= 0 && y < 32 && xx >= 0 && xx < 32) ? in32[y][xx] : 0.f;
        acc += w[cc * 9 + ky * 3 + kx] * vv;
      }
    t[cc][p] = acc >= 0.f ? acc : 0.2f * acc;   // leaky relu
  }
  __syncthreads();

  for (int cc = 0; cc < 4; ++cc) {
    red[p] = t[cc][p];
    __syncthreads();
    for (int s = 128; s > 0; s >>= 1) { if (p < s) red[p] += red[p + s]; __syncthreads(); }
    float m = red[0] * (1.f / 256.f);
    __syncthreads();
    float dlt = t[cc][p] - m;
    red[p] = dlt * dlt;
    __syncthreads();
    for (int s = 128; s > 0; s >>= 1) { if (p < s) red[p] += red[p + s]; __syncthreads(); }
    float var = red[0] * (1.f / 256.f);
    __syncthreads();
    out[((size_t)b * 4 + cc) * 256 + p] = dlt * rsqrtf(var + 1e-5f) * g[cc] + bt[cc];
  }
}

// ------- kernel 4: attention + MLP + LN + 16x16 "conv" -> level[b] -------
__global__ __launch_bounds__(256) void attention_kernel(
    const float* __restrict__ v32n, const float* __restrict__ qfn,
    const float* Wq, const float* Wk, const float* Wv, const float* Wfc,
    const float* W1, const float* b1, const float* W2, const float* b2,
    const float* lng, const float* lnb, const float* wF, const float* bF,
    float* __restrict__ level) {
  const int b = blockIdx.x, p = threadIdx.x;
  __shared__ float kL[4][256], qL[4][256], vL[4][256], aL[4][256], oL[4][256];
  __shared__ float mx[4][16], sm[4][16];
  __shared__ float red[256];

  float vv[4], qv[4];
#pragma unroll
  for (int c = 0; c < 4; ++c) {
    vv[c] = v32n[((size_t)b * 4 + c) * 256 + p];
    qv[c] = qfn[((size_t)b * 4 + c) * 256 + p];
  }
#pragma unroll
  for (int cc = 0; cc < 4; ++cc) {
    float sk = 0, sq = 0, sv = 0;
#pragma unroll
    for (int c = 0; c < 4; ++c) {
      sk += Wk[cc * 4 + c] * vv[c];
      sq += Wq[cc * 4 + c] * qv[c];
      sv += Wv[cc * 4 + c] * vv[c];
    }
    kL[cc][p] = sk; qL[cc][p] = sq; vL[cc][p] = sv;
  }
  __syncthreads();

  const int i = p >> 4, j = p & 15;
  for (int c = 0; c < 4; ++c) {
    float s = 0;
#pragma unroll
    for (int t = 0; t < 16; ++t) s += kL[c][i * 16 + t] * qL[c][t * 16 + j];
    aL[c][p] = s;
  }
  __syncthreads();
  if (p < 64) {
    int c = p >> 4, ii = p & 15;
    float m = -1e30f;
    for (int jj = 0; jj < 16; ++jj) m = fmaxf(m, aL[c][ii * 16 + jj]);
    float s = 0;
    for (int jj = 0; jj < 16; ++jj) s += expf(aL[c][ii * 16 + jj] - m);
    mx[c][ii] = m; sm[c][ii] = s;
  }
  __syncthreads();
  for (int c = 0; c < 4; ++c) aL[c][p] = expf(aL[c][p] - mx[c][i]) / sm[c][i];
  __syncthreads();
  for (int c = 0; c < 4; ++c) {
    float s = 0;
#pragma unroll
    for (int t = 0; t < 16; ++t) s += vL[c][i * 16 + t] * aL[c][t * 16 + j];
    oL[c][p] = s;
  }
  __syncthreads();

  float a4[4], afc[4], at[4];
#pragma unroll
  for (int c = 0; c < 4; ++c) a4[c] = oL[c][p];
#pragma unroll
  for (int cc = 0; cc < 4; ++cc) {
    float s = 0;
#pragma unroll
    for (int c = 0; c < 4; ++c) s += Wfc[cc * 4 + c] * a4[c];
    afc[cc] = s;
  }
  {
    float a2[4] = {0, 0, 0, 0};
    for (int jj = 0; jj < 32; ++jj) {
      float s = b1[jj];
#pragma unroll
      for (int c = 0; c < 4; ++c) s += W1[jj * 4 + c] * afc[c];
      float h = s > 0.f ? s : 0.f;
#pragma unroll
      for (int cc = 0; cc < 4; ++cc) a2[cc] += W2[cc * 32 + jj] * h;
    }
#pragma unroll
    for (int cc = 0; cc < 4; ++cc) at[cc] = afc[cc] + a2[cc] + b2[cc];
  }
  float mu = (at[0] + at[1] + at[2] + at[3]) * 0.25f;
  float var = 0;
#pragma unroll
  for (int c = 0; c < 4; ++c) { float dd = at[c] - mu; var += dd * dd; }
  var *= 0.25f;
  float rs = rsqrtf(var + 1e-5f);
#pragma unroll
  for (int c = 0; c < 4; ++c) oL[c][p] = (at[c] - mu) * rs * lng[c] + lnb[c];  // v5
  __syncthreads();

  float part = 0;
#pragma unroll
  for (int c = 0; c < 4; ++c) part += wF[c * 256 + p] * oL[c][p];
  red[p] = part;
  __syncthreads();
  for (int s = 128; s > 0; s >>= 1) { if (p < s) red[p] += red[p + s]; __syncthreads(); }
  if (p == 0) {
    float lv = red[0] + bF[0];
    float sg = 1.f / (1.f + expf(-lv));
    level[b] = 0.5f * sg + 0.3f;
  }
}

// ---------------- kernel 5: final per-pixel enhancement ----------------
__global__ __launch_bounds__(256) void final_kernel(
    const float* __restrict__ x, const float* __restrict__ vbuf,
    const float* __restrict__ rbuf, const float* __restrict__ level,
    float* __restrict__ out) {
  size_t p = (size_t)blockIdx.x * 256 + threadIdx.x;    // over B*HW
  int b = (int)(p >> 18);
  size_t rem = p & (size_t)(HWP - 1);
  float v = vbuf[p];
  float r = rbuf[p];
  float ev1 = v + v * r / (1.f - r + 1e-6f);
  float vvv = v + 1e-6f;
  float lv = level[b];
  size_t base = (size_t)b * 3 * HWP + rem;
#pragma unroll
  for (int c = 0; c < 3; ++c) {
    float ch = x[base + (size_t)c * HWP];
    float x0 = ch / vvv * ev1;
    x0 = fminf(fmaxf(x0, 1e-6f), 0.999999f);
    out[base + (size_t)c * HWP] = expf(lv * logf(x0));  // pow(x0, level)
  }
}

// ---------------- host launcher ----------------
extern "C" void kernel_launch(void* const* d_in, const int* in_sizes, int n_in,
                              void* d_out, int out_size, void* d_ws, size_t ws_size,
                              hipStream_t stream) {
  (void)in_sizes; (void)n_in; (void)out_size; (void)ws_size;
  const float* x = (const float*)d_in[0];
  auto F = [&](int i) { return (const float*)d_in[i]; };

  float* out   = (float*)d_out;
  float* r_out = out + (size_t)BB * 3 * HWP;            // second output slab

  float*    ws    = (float*)d_ws;
  float*    v_f32 = ws;                                  // B*HW floats
  _Float16* v_f16 = (_Float16*)(ws + (size_t)BB * HWP);  // B*HW halfs
  float*    v32n  = ws + (size_t)BB * HWP + (size_t)BB * HWP / 2;
  float*    qfn   = v32n + BB * 4 * 256;
  float*    level = qfn + BB * 4 * 256;

  // 1) v = mean(rgb)
  prep_v_kernel<<<(BB * HWP) / 256, 256, 0, stream>>>(x, v_f32, v_f16);

  // 2) fused UNet++ -> r (sigmoid), WMMA in LDS, one 32x32 tile per workgroup
  unet_tile_kernel<<<dim3(WWID / TILE, HH / TILE, BB), 256, SMEM_BYTES, stream>>>(
      v_f16,
      F(1), F(2), F(3), F(4), F(5), F(6), F(7), F(8), F(9), F(10),
      F(11), F(12), F(13), F(14), F(15), F(16), F(17), F(18), F(19), F(20),
      r_out);

  // 3) downsample + stride-2 conv + lrelu + instnorm (v path, then r path)
  smallmap_kernel<<<BB, 256, 0, stream>>>(v_f32, F(21), F(22), F(23), F(24), v32n);
  smallmap_kernel<<<BB, 256, 0, stream>>>(r_out, F(25), F(26), F(27), F(28), qfn);

  // 4) attention + MLP + LN + level conv
  attention_kernel<<<BB, 256, 0, stream>>>(v32n, qfn,
      F(31), F(32), F(33), F(34), F(35), F(36), F(37), F(38), F(39), F(40),
      F(29), F(30), level);

  // 5) final per-pixel enhancement
  final_kernel<<<(BB * HWP) / 256, 256, 0, stream>>>(x, v_f32, r_out, level, out);
}
